// CWRRTEWindowCell_38689065402923
// MI455X (gfx1250) — matrix operand order repaired and checked by reading, not compiled, not run on hardware
//
#include <hip/hip_runtime.h>

typedef unsigned short u16;
typedef __attribute__((ext_vector_type(16))) __bf16 v16bf;
typedef __attribute__((ext_vector_type(8)))  float  v8f;

#define Bn   32
#define Tn   512
#define Dn   512
#define Hn   8
#define HDn  64
#define Ln   6
#define FFn  2048
#define Mrows (Bn*Tn)   // 16384
#define EHn  4
#define EHDn 128

union AFrag { v16bf v; uint4 q[2]; u16 u[16]; };
union CFrag { v8f   v; float f[8]; };

__device__ __forceinline__ u16 f2bf(float f) {
    union { float f; unsigned int u; } c; c.f = f;
    unsigned int u = c.u;
    unsigned int r = u + 0x7FFFu + ((u >> 16) & 1u);
    return (u16)(r >> 16);
}
__device__ __forceinline__ float bf2f(u16 h) {
    union { unsigned int u; float f; } c; c.u = ((unsigned int)h) << 16;
    return c.f;
}
__device__ __forceinline__ v8f wmma_bf16(v16bf a, v16bf b, v8f c) {
    return __builtin_amdgcn_wmma_f32_16x16x32_bf16(false, a, false, b, (short)0, c, false, false);
}

// ---------------- weight transpose + convert: Wt[n*K+k] = bf16(W[k*ldW + coloff + n])
__global__ void k_transpose_w(const float* __restrict__ W, u16* __restrict__ Wt,
                              int K, int N, int ldW, int coloff) {
    size_t tid = (size_t)blockIdx.x * blockDim.x + threadIdx.x;
    if (tid >= (size_t)K * N) return;
    int k = (int)(tid % K);
    int n = (int)(tid / K);
    Wt[(size_t)n * K + k] = f2bf(W[(size_t)k * ldW + coloff + n]);
}

// ---------------- ctx projection: ctxb[b][d] = concat(fast,slow) @ ctx_W + ctx_b
__global__ void k_ctx_proj(const float* __restrict__ fast, const float* __restrict__ slow,
                           const float* __restrict__ W, const float* __restrict__ bias,
                           float* __restrict__ ctxb) {
    int tid = blockIdx.x * blockDim.x + threadIdx.x;  // Bn*Dn
    int d = tid & (Dn - 1);
    int b = tid >> 9;
    float acc = bias[d];
    const float* f = fast + b * Dn;
    const float* s = slow + b * Dn;
    for (int j = 0; j < Dn; j++) acc += f[j] * W[(size_t)j * Dn + d];
    for (int j = 0; j < Dn; j++) acc += s[j] * W[(size_t)(Dn + j) * Dn + d];
    ctxb[tid] = acc;
}

// ---------------- engram hash indices
__global__ void k_engram_idx(const int* __restrict__ tokens, const int* __restrict__ prev,
                             int* __restrict__ idx) {
    int tid = blockIdx.x * blockDim.x + threadIdx.x;  // Bn*Tn*EHn
    int e = tid & 3;
    int t = (tid >> 2) & (Tn - 1);
    int b = tid >> 11;
    unsigned x = 131u + 1009u * (unsigned)e;
    unsigned pr[4];
    for (int j = 0; j < 4; j++) { pr[j] = x; x = x * 31u + 1u; }
    unsigned hs = 0u;
    for (int i = 0; i < 4; i++) {
        int pos = 64 - i + t;
        unsigned tok = (pos >= 64) ? (unsigned)tokens[b * Tn + (pos - 64)]
                                   : (unsigned)prev[b * 64 + pos];
        hs += tok * pr[i];
    }
    idx[tid] = (int)(hs % 100000u);
}

// ---------------- x init: embed + ctx + engram
__global__ void k_init_x(const int* __restrict__ tokens, const float* __restrict__ embed,
                         const float* __restrict__ ctxb, const int* __restrict__ idx,
                         const float* __restrict__ etab, const float* __restrict__ egate,
                         float* __restrict__ x) {
    size_t tid = (size_t)blockIdx.x * blockDim.x + threadIdx.x;  // Bn*Tn*Dn
    int d = (int)(tid & (Dn - 1));
    size_t bt = tid >> 9;
    int b = (int)(bt >> 9);
    int e = d >> 7, dd = d & 127;
    int tok = tokens[bt];
    int ix = idx[bt * 4 + e];
    float g = 1.f / (1.f + expf(-egate[e * EHDn + dd]));
    x[tid] = embed[(size_t)tok * Dn + d] + ctxb[b * Dn + d]
           + etab[((size_t)ix * EHn + e) * EHDn + dd] * g;
}

// ---------------- RMSNorm: one wave per row of 512; out bf16 or f32 (in-place ok)
__global__ void k_rmsnorm(const float* __restrict__ x, const float* __restrict__ gamma,
                          u16* outb, float* outf) {
    int wave = threadIdx.x >> 5, lane = threadIdx.x & 31;
    size_t row = (size_t)blockIdx.x * 8 + wave;
    const float* xr = x + row * Dn;
    float ss = 0.f;
    float v[16];
    for (int i = 0; i < 16; i++) { v[i] = xr[lane + i * 32]; ss += v[i] * v[i]; }
    for (int m = 1; m < 32; m <<= 1) ss += __shfl_xor(ss, m, 32);
    float inv = rsqrtf(ss / (float)Dn + 1e-6f);
    for (int i = 0; i < 16; i++) {
        int c = lane + i * 32;
        float o = v[i] * inv * gamma[c];
        if (outb) outb[row * Dn + c] = f2bf(o);
        else      outf[row * Dn + c] = o;
    }
}

// ---------------- WMMA GEMM: C[M,N] = A[M,K](bf16,row) @ Bt[N,K](bf16)ᵀ (+bias)
// mode 0: outb = bf16(acc+bias)
// mode 1: outf += acc+bias
// mode 2: outb = bf16( silu(bf2f(prevb)) * acc )   (SwiGLU combine)
__global__ void k_gemm_wmma(const u16* __restrict__ A, int lda,
                            const u16* __restrict__ Bt, int ldb,
                            const float* __restrict__ bias,
                            u16* outb, float* outf, const u16* prevb,
                            int ldc, int K, int mode) {
    int wave = threadIdx.x >> 5, lane = threadIdx.x & 31;
    int lhi = lane >> 4, llo = lane & 15;
    int m0 = blockIdx.y * 128 + (wave & 3) * 32;
    int n0 = blockIdx.x * 64 + (wave >> 2) * 32;
    CFrag acc[2][2];
    for (int a = 0; a < 2; a++) for (int b = 0; b < 2; b++)
        for (int r = 0; r < 8; r++) acc[a][b].f[r] = 0.f;

    for (int kc = 0; kc < K; kc += 32) {
        AFrag af[2], bf[2];
        for (int mi = 0; mi < 2; mi++) {
            const u16* ar = A + (size_t)(m0 + mi * 16 + llo) * lda + kc + lhi * 8;
            af[mi].q[0] = *(const uint4*)(ar);
            af[mi].q[1] = *(const uint4*)(ar + 16);
        }
        for (int ni = 0; ni < 2; ni++) {
            const u16* br = Bt + (size_t)(n0 + ni * 16 + llo) * ldb + kc + lhi * 8;
            bf[ni].q[0] = *(const uint4*)(br);
            bf[ni].q[1] = *(const uint4*)(br + 16);
        }
        for (int mi = 0; mi < 2; mi++)
            for (int ni = 0; ni < 2; ni++)
                acc[mi][ni].v = wmma_bf16(af[mi].v, bf[ni].v, acc[mi][ni].v);
    }
    for (int mi = 0; mi < 2; mi++)
        for (int ni = 0; ni < 2; ni++)
            for (int r = 0; r < 8; r++) {
                int m = m0 + mi * 16 + r + lhi * 8;
                int n = n0 + ni * 16 + llo;
                float v = acc[mi][ni].f[r] + (bias ? bias[n] : 0.f);
                size_t off = (size_t)m * ldc + n;
                if (mode == 0) outb[off] = f2bf(v);
                else if (mode == 1) outf[off] += v;
                else {
                    float g = bf2f(prevb[off]);
                    float sg = g / (1.f + expf(-g));
                    outb[off] = f2bf(sg * v);
                }
            }
}

// ---------------- RoPE pack: q->(B,H,T,HD)*scl, k->(B,H,T,HD), v->(B,H,HD,T)
__global__ void k_rope_pack(const u16* __restrict__ qf, const u16* __restrict__ kf,
                            const u16* __restrict__ vf,
                            u16* __restrict__ qb, u16* __restrict__ kb, u16* __restrict__ vt) {
    int tid = blockIdx.x * blockDim.x + threadIdx.x;  // Bn*Tn*Hn*32
    int i = tid & 31;
    int h = (tid >> 5) & 7;
    int t = (tid >> 8) & (Tn - 1);
    int b = tid >> 17;
    size_t src = ((size_t)(b * Tn + t)) * Dn + h * HDn;
    float q0 = bf2f(qf[src + 2 * i]), q1 = bf2f(qf[src + 2 * i + 1]);
    float k0 = bf2f(kf[src + 2 * i]), k1 = bf2f(kf[src + 2 * i + 1]);
    float ang = (float)t * powf(10000.f, -(float)(2 * i) / (float)HDn);
    float cs = cosf(ang), sn = sinf(ang);
    const float scl = 0.125f;  // 1/sqrt(64)
    size_t dst = ((size_t)(b * Hn + h) * Tn + t) * HDn;
    qb[dst + 2 * i]     = f2bf((q0 * cs - q1 * sn) * scl);
    qb[dst + 2 * i + 1] = f2bf((q0 * sn + q1 * cs) * scl);
    kb[dst + 2 * i]     = f2bf(k0 * cs - k1 * sn);
    kb[dst + 2 * i + 1] = f2bf(k0 * sn + k1 * cs);
    size_t vbase = ((size_t)(b * Hn + h) * HDn);
    vt[(vbase + 2 * i) * Tn + t]     = vf[src + 2 * i];
    vt[(vbase + 2 * i + 1) * Tn + t] = vf[src + 2 * i + 1];
}

// ---------------- Flash attention with WMMA: one wave = 16-query tile
__global__ void k_attn(const u16* __restrict__ Qb, const u16* __restrict__ Kb,
                       const u16* __restrict__ Vt, u16* __restrict__ attnb) {
    __shared__ __align__(16) u16 Pt[8][16 * 32];
    int wave = threadIdx.x >> 5, lane = threadIdx.x & 31;
    int lhi = lane >> 4, llo = lane & 15;
    int blk = blockIdx.x;                 // Bn*Hn*4
    int qt = blk & 3;
    int bh = blk >> 2;
    int h = bh & 7, b = bh >> 3;
    int qbase = qt * 128 + wave * 16;

    const u16* Q  = Qb + (size_t)bh * Tn * HDn;
    const u16* Kp = Kb + (size_t)bh * Tn * HDn;
    const u16* Vp = Vt + (size_t)bh * HDn * Tn;

    AFrag aq[2];
    {
        const u16* qrow = Q + (size_t)(qbase + llo) * HDn + lhi * 8;
        for (int c = 0; c < 2; c++) {
            aq[c].q[0] = *(const uint4*)(qrow + c * 32);
            aq[c].q[1] = *(const uint4*)(qrow + c * 32 + 16);
        }
    }
    CFrag o[4];
    for (int nt = 0; nt < 4; nt++) for (int r = 0; r < 8; r++) o[nt].f[r] = 0.f;
    float mrun[8], lrun[8];
    for (int r = 0; r < 8; r++) { mrun[r] = -1e30f; lrun[r] = 0.f; }

    int kbmax = (qbase + 15) >> 5;
    for (int jb = 0; jb <= kbmax; jb++) {
        int j0 = jb * 32;
        CFrag s[2];
        for (int half = 0; half < 2; half++) {
            for (int r = 0; r < 8; r++) s[half].f[r] = 0.f;
            const u16* krow = Kp + (size_t)(j0 + half * 16 + llo) * HDn + lhi * 8;
            for (int c = 0; c < 2; c++) {
                AFrag bk;
                bk.q[0] = *(const uint4*)(krow + c * 32);
                bk.q[1] = *(const uint4*)(krow + c * 32 + 16);
                s[half].v = wmma_bf16(aq[c].v, bk.v, s[half].v);
            }
        }
        // causal mask
        for (int r = 0; r < 8; r++) {
            int qrow = qbase + r + lhi * 8;
            int k0i = j0 + llo, k1i = k0i + 16;
            if (k0i > qrow) s[0].f[r] = -1e9f;
            if (k1i > qrow) s[1].f[r] = -1e9f;
        }
        // online softmax
        float p0[8], p1[8];
        for (int r = 0; r < 8; r++) {
            float rmax = fmaxf(s[0].f[r], s[1].f[r]);
            for (int m = 1; m < 16; m <<= 1) rmax = fmaxf(rmax, __shfl_xor(rmax, m, 32));
            float mnew = fmaxf(mrun[r], rmax);
            float alpha = expf(mrun[r] - mnew);
            p0[r] = expf(s[0].f[r] - mnew);
            p1[r] = expf(s[1].f[r] - mnew);
            float rs = p0[r] + p1[r];
            for (int m = 1; m < 16; m <<= 1) rs += __shfl_xor(rs, m, 32);
            lrun[r] = lrun[r] * alpha + rs;
            mrun[r] = mnew;
            for (int nt = 0; nt < 4; nt++) o[nt].f[r] *= alpha;
        }
        // stage P (16x32 bf16) to LDS to re-layout C-frag -> A-frag
        for (int r = 0; r < 8; r++) {
            int row = r + lhi * 8;
            Pt[wave][row * 32 + llo]      = f2bf(p0[r]);
            Pt[wave][row * 32 + 16 + llo] = f2bf(p1[r]);
        }
        AFrag aP;
        {
            const u16* prow = &Pt[wave][llo * 32 + lhi * 8];
            aP.q[0] = *(const uint4*)(prow);
            aP.q[1] = *(const uint4*)(prow + 16);
        }
        // O += P @ V
        for (int nt = 0; nt < 4; nt++) {
            const u16* vrow = Vp + (size_t)(nt * 16 + llo) * Tn + j0 + lhi * 8;
            AFrag bv;
            bv.q[0] = *(const uint4*)(vrow);
            bv.q[1] = *(const uint4*)(vrow + 16);
            o[nt].v = wmma_bf16(aP.v, bv.v, o[nt].v);
        }
    }
    for (int nt = 0; nt < 4; nt++)
        for (int r = 0; r < 8; r++) {
            int qrow = qbase + r + lhi * 8;
            int col = h * HDn + nt * 16 + llo;
            attnb[((size_t)(b * Tn + qrow)) * Dn + col] = f2bf(o[nt].f[r] / lrun[r]);
        }
}

// ---------------- salience pooling: wv[b,e,:] = softmax_t(logits) weighted sum
__global__ void k_salience(const float* __restrict__ x, const int* __restrict__ tokens,
                           const float* __restrict__ salW, const float* __restrict__ salb,
                           const float* __restrict__ temp, float* __restrict__ wv) {
    __shared__ float sh[Tn];
    __shared__ float red[128];
    int e = blockIdx.x & 3, b = blockIdx.x >> 2;
    float tpr = log1pf(expf(temp[e])) + 0.3f;
    for (int t = threadIdx.x; t < Tn; t += 128) {
        const float* xr = x + ((size_t)(b * Tn + t)) * Dn;
        float acc = 0.f;
        for (int j = 0; j < Dn; j++) acc += xr[j] * salW[j * EHn + e];
        float lg = (acc + salb[e]) / tpr;
        sh[t] = (tokens[b * Tn + t] != 0) ? lg : -1e9f;
    }
    __syncthreads();
    float mx = -1e30f;
    for (int t = threadIdx.x; t < Tn; t += 128) mx = fmaxf(mx, sh[t]);
    red[threadIdx.x] = mx; __syncthreads();
    for (int s2 = 64; s2 > 0; s2 >>= 1) {
        if (threadIdx.x < s2) red[threadIdx.x] = fmaxf(red[threadIdx.x], red[threadIdx.x + s2]);
        __syncthreads();
    }
    mx = red[0]; __syncthreads();
    float sm = 0.f;
    for (int t = threadIdx.x; t < Tn; t += 128) {
        float ev = (tokens[b * Tn + t] != 0) ? expf(sh[t] - mx) : 0.f;
        sh[t] = ev; sm += ev;
    }
    __syncthreads();
    red[threadIdx.x] = sm; __syncthreads();
    for (int s2 = 64; s2 > 0; s2 >>= 1) {
        if (threadIdx.x < s2) red[threadIdx.x] += red[threadIdx.x + s2];
        __syncthreads();
    }
    float denom = red[0] + 1e-6f;
    __syncthreads();
    float acc = 0.f;
    for (int t = 0; t < Tn; t++)
        acc += sh[t] * x[((size_t)(b * Tn + t)) * Dn + e * EHDn + threadIdx.x];
    wv[(size_t)(b * EHn + e) * EHDn + threadIdx.x] = acc / denom;
}

// ---------------- state update: gate + rmsnorm(write_vec) + EMA
__global__ void k_state_update(const float* __restrict__ wv, const float* __restrict__ gateW,
                               const float* __restrict__ gateb, const float* __restrict__ salrms,
                               const int* __restrict__ tokens,
                               const float* __restrict__ fast, const float* __restrict__ slow,
                               float* __restrict__ ofast, float* __restrict__ oslow) {
    __shared__ float shred[256];
    __shared__ float gl[4];
    __shared__ int anyv;
    int b = blockIdx.x;
    const float* w = wv + (size_t)b * Dn;
    float ss = 0.f;
    for (int d = threadIdx.x; d < Dn; d += 256) { float v = w[d]; ss += v * v; }
    shred[threadIdx.x] = ss;
    if (threadIdx.x == 0) anyv = 0;
    __syncthreads();
    for (int s = 128; s > 0; s >>= 1) {
        if (threadIdx.x < s) shred[threadIdx.x] += shred[threadIdx.x + s];
        __syncthreads();
    }
    float inv = rsqrtf(shred[0] / (float)Dn + 1e-6f);
    for (int t = threadIdx.x; t < Tn; t += 256)
        if (tokens[b * Tn + t] != 0) anyv = 1;
    if (threadIdx.x < 4) {
        float acc = gateb[0];
        for (int j = 0; j < EHDn; j++) acc += w[threadIdx.x * EHDn + j] * gateW[j];
        gl[threadIdx.x] = acc;
    }
    __syncthreads();
    float valid = anyv ? 1.f : 0.f;
    for (int d = threadIdx.x; d < Dn; d += 256) {
        int e = d >> 7;
        float u = (1.f / (1.f + expf(-gl[e]))) * valid;
        float wvn = w[d] * inv * salrms[d];
        ofast[b * Dn + d] = (1.f - u) * fast[b * Dn + d] + u * wvn;
        oslow[b * Dn + d] = (1.f - 0.1f * u) * slow[b * Dn + d] + 0.1f * u * wvn;
    }
}

extern "C" void kernel_launch(void* const* d_in, const int* in_sizes, int n_in,
                              void* d_out, int out_size, void* d_ws, size_t ws_size,
                              hipStream_t stream) {
    const int*   tokens = (const int*)  d_in[0];
    const int*   prev   = (const int*)  d_in[1];
    const float* ssf    = (const float*)d_in[2];
    const float* sss    = (const float*)d_in[3];
    const float* embed  = (const float*)d_in[4];
    const float* ctxW   = (const float*)d_in[5];
    const float* ctxB   = (const float*)d_in[6];
    const float* etab   = (const float*)d_in[7];
    const float* egate  = (const float*)d_in[8];
    const float* ln1    = (const float*)d_in[9];
    const float* Wq     = (const float*)d_in[10];
    const float* bq     = (const float*)d_in[11];
    const float* Wk     = (const float*)d_in[12];
    const float* bk     = (const float*)d_in[13];
    const float* Wv     = (const float*)d_in[14];
    const float* bv     = (const float*)d_in[15];
    const float* Wo     = (const float*)d_in[16];
    const float* bo     = (const float*)d_in[17];
    const float* ln2    = (const float*)d_in[18];
    const float* W1     = (const float*)d_in[19];
    const float* W2     = (const float*)d_in[20];
    const float* lnf    = (const float*)d_in[21];
    const float* salW   = (const float*)d_in[22];
    const float* salb   = (const float*)d_in[23];
    const float* temp   = (const float*)d_in[24];
    const float* gateW  = (const float*)d_in[25];
    const float* gateb  = (const float*)d_in[26];
    const float* salrms = (const float*)d_in[27];

    float* x     = (float*)d_out;                       // (B,T,D)
    float* ofast = x + (size_t)Mrows * Dn;              // (B,D)
    float* oslow = ofast + (size_t)Bn * Dn;             // (B,D)

    // workspace layout
    char* base = (char*)d_ws;
    size_t off = 0;
    auto alloc = [&](size_t bytes) -> char* {
        char* p = base + off;
        off = (off + bytes + 255) & ~(size_t)255;
        return p;
    };
    const size_t szDD  = (size_t)Dn * Dn * 2;           // 512x512 bf16
    u16* wqT  = (u16*)alloc(szDD * Ln);
    u16* wkT  = (u16*)alloc(szDD * Ln);
    u16* wvT  = (u16*)alloc(szDD * Ln);
    u16* woT  = (u16*)alloc(szDD * Ln);
    u16* w1aT = (u16*)alloc((size_t)FFn * Dn * 2 * Ln);
    u16* w1bT = (u16*)alloc((size_t)FFn * Dn * 2 * Ln);
    u16* w2T  = (u16*)alloc((size_t)Dn * FFn * 2 * Ln);
    u16* nx    = (u16*)alloc((size_t)Mrows * Dn * 2);
    u16* qf    = (u16*)alloc((size_t)Mrows * Dn * 2);
    u16* kf    = (u16*)alloc((size_t)Mrows * Dn * 2);
    u16* vf    = (u16*)alloc((size_t)Mrows * Dn * 2);
    u16* qb    = (u16*)alloc((size_t)Mrows * Dn * 2);
    u16* kb    = (u16*)alloc((size_t)Mrows * Dn * 2);
    u16* vt    = (u16*)alloc((size_t)Mrows * Dn * 2);
    u16* attnb = (u16*)alloc((size_t)Mrows * Dn * 2);
    u16* hbuf  = (u16*)alloc((size_t)Mrows * FFn * 2);
    float* ctxb = (float*)alloc((size_t)Bn * Dn * 4);
    int*   idxb = (int*)  alloc((size_t)Bn * Tn * EHn * 4);
    float* wvb  = (float*)alloc((size_t)Bn * EHn * EHDn * 4);
    (void)ws_size; (void)n_in; (void)in_sizes; (void)out_size;

    // --- weight conversion (transpose to N x K bf16) ---
    for (int l = 0; l < Ln; l++) {
        int gDD = (Dn * Dn) / 256;
        k_transpose_w<<<gDD, 256, 0, stream>>>(Wq + (size_t)l*Dn*Dn, wqT + (size_t)l*Dn*Dn, Dn, Dn, Dn, 0);
        k_transpose_w<<<gDD, 256, 0, stream>>>(Wk + (size_t)l*Dn*Dn, wkT + (size_t)l*Dn*Dn, Dn, Dn, Dn, 0);
        k_transpose_w<<<gDD, 256, 0, stream>>>(Wv + (size_t)l*Dn*Dn, wvT + (size_t)l*Dn*Dn, Dn, Dn, Dn, 0);
        k_transpose_w<<<gDD, 256, 0, stream>>>(Wo + (size_t)l*Dn*Dn, woT + (size_t)l*Dn*Dn, Dn, Dn, Dn, 0);
        int gDF = (Dn * FFn) / 256;
        k_transpose_w<<<gDF, 256, 0, stream>>>(W1 + (size_t)l*Dn*2*FFn, w1aT + (size_t)l*FFn*Dn, Dn, FFn, 2*FFn, 0);
        k_transpose_w<<<gDF, 256, 0, stream>>>(W1 + (size_t)l*Dn*2*FFn, w1bT + (size_t)l*FFn*Dn, Dn, FFn, 2*FFn, FFn);
        k_transpose_w<<<gDF, 256, 0, stream>>>(W2 + (size_t)l*FFn*Dn,  w2T  + (size_t)l*Dn*FFn, FFn, Dn, Dn, 0);
    }

    // --- front-end ---
    k_ctx_proj<<<(Bn * Dn) / 256, 256, 0, stream>>>(ssf, sss, ctxW, ctxB, ctxb);
    k_engram_idx<<<(Bn * Tn * EHn) / 256, 256, 0, stream>>>(tokens, prev, idxb);
    k_init_x<<<(Mrows * Dn) / 256, 256, 0, stream>>>(tokens, embed, ctxb, idxb, etab, egate, x);

    dim3 gProj(Dn / 64, Mrows / 128);   // N=512
    dim3 gFF(FFn / 64, Mrows / 128);    // N=2048

    for (int l = 0; l < Ln; l++) {
        const u16* wq_l = wqT + (size_t)l * Dn * Dn;
        const u16* wk_l = wkT + (size_t)l * Dn * Dn;
        const u16* wv_l = wvT + (size_t)l * Dn * Dn;
        const u16* wo_l = woT + (size_t)l * Dn * Dn;
        const u16* w1a_l = w1aT + (size_t)l * FFn * Dn;
        const u16* w1b_l = w1bT + (size_t)l * FFn * Dn;
        const u16* w2_l  = w2T  + (size_t)l * Dn * FFn;

        k_rmsnorm<<<Mrows / 8, 256, 0, stream>>>(x, ln1 + l * Dn, nx, nullptr);

        k_gemm_wmma<<<gProj, 256, 0, stream>>>(nx, Dn, wq_l, Dn, bq + l * Dn, qf, nullptr, nullptr, Dn, Dn, 0);
        k_gemm_wmma<<<gProj, 256, 0, stream>>>(nx, Dn, wk_l, Dn, bk + l * Dn, kf, nullptr, nullptr, Dn, Dn, 0);
        k_gemm_wmma<<<gProj, 256, 0, stream>>>(nx, Dn, wv_l, Dn, bv + l * Dn, vf, nullptr, nullptr, Dn, Dn, 0);

        k_rope_pack<<<(Bn * Tn * Hn * 32) / 256, 256, 0, stream>>>(qf, kf, vf, qb, kb, vt);

        k_attn<<<Bn * Hn * 4, 256, 0, stream>>>(qb, kb, vt, attnb);

        k_gemm_wmma<<<gProj, 256, 0, stream>>>(attnb, Dn, wo_l, Dn, bo + l * Dn, nullptr, x, nullptr, Dn, Dn, 1);

        k_rmsnorm<<<Mrows / 8, 256, 0, stream>>>(x, ln2 + l * Dn, nx, nullptr);

        k_gemm_wmma<<<gFF, 256, 0, stream>>>(nx, Dn, w1a_l, Dn, nullptr, hbuf, nullptr, nullptr, FFn, Dn, 0);
        k_gemm_wmma<<<gFF, 256, 0, stream>>>(nx, Dn, w1b_l, Dn, nullptr, hbuf, nullptr, hbuf, FFn, Dn, 2);
        k_gemm_wmma<<<gProj, 256, 0, stream>>>(hbuf, FFn, w2_l, FFn, nullptr, nullptr, x, nullptr, Dn, FFn, 1);
    }

    k_rmsnorm<<<Mrows / 8, 256, 0, stream>>>(x, lnf, nullptr, x);  // in-place final norm

    k_salience<<<Bn * EHn, 128, 0, stream>>>(x, tokens, salW, salb, temp, wvb);
    k_state_update<<<Bn, 256, 0, stream>>>(wvb, gateW, gateb, salrms, tokens, ssf, sss, ofast, oslow);
}
